// DoubleGCNnoBN_49228915147575
// MI455X (gfx1250) — matrix operand order repaired and checked by reading, not compiled, not run on hardware
//
#include <hip/hip_runtime.h>

// ---------- types / helpers ----------
typedef __attribute__((ext_vector_type(16))) __bf16 v16bf;
typedef __attribute__((ext_vector_type(8)))  float  v8f;

union FragA { v16bf v; unsigned short u[16]; uint4 q[2]; };

__device__ __forceinline__ unsigned short f2bf(float f) {
  unsigned int u = __float_as_uint(f);
  unsigned int r = (u + 0x7FFFu + ((u >> 16) & 1u)) >> 16;  // RNE
  return (unsigned short)r;
}
__device__ __forceinline__ float bf2f(unsigned short s) {
  return __uint_as_float(((unsigned int)s) << 16);
}

// ---------- utility kernels ----------
__global__ void k_zero_f32(float* p, long long n) {
  long long i = (long long)blockIdx.x * blockDim.x + threadIdx.x;
  if (i < n) p[i] = 0.0f;
}

// vectorized fp32 -> bf16 streaming convert (4 elements/thread)
__global__ void k_conv_bf16_vec4(const float* __restrict__ in, unsigned short* __restrict__ out,
                                 long long n4) {
  long long i = (long long)blockIdx.x * blockDim.x + threadIdx.x;
  if (i >= n4) return;
  float4 f = ((const float4*)in)[i];
  ushort4 o;
  o.x = f2bf(f.x); o.y = f2bf(f.y); o.z = f2bf(f.z); o.w = f2bf(f.w);
  ((ushort4*)out)[i] = o;
}

__global__ void k_count_deg(const int* __restrict__ idx, float* __restrict__ deg, int nE) {
  int i = blockIdx.x * blockDim.x + threadIdx.x;
  if (i < nE) atomicAdd(&deg[idx[i]], 1.0f);
}

__global__ void k_norm_coef(float* c, int n) {
  int i = blockIdx.x * blockDim.x + threadIdx.x;
  if (i < n) c[i] = rsqrtf(fmaxf(c[i], 1.0f));
}

// Wt[n*K + k] = bf16(W[k*Nc + n])  (transpose so B fragments are contiguous)
__global__ void k_convT(const float* __restrict__ W, unsigned short* __restrict__ Wt,
                        int K, int Nc) {
  long long i = (long long)blockIdx.x * blockDim.x + threadIdx.x;
  if (i < (long long)K * Nc) {
    int n = (int)(i / K), k = (int)(i - (long long)n * K);
    Wt[i] = f2bf(W[(size_t)k * Nc + n]);
  }
}

// ---------- WMMA GEMM: out[M,Nc](bf16) = in[M,K](bf16) @ Wt^T ; one wave = 16x64 tile ------
__global__ void k_gemm_wmma(const unsigned short* __restrict__ inp,
                            const unsigned short* __restrict__ Wt,
                            unsigned short* __restrict__ outp, int M, int K, int Nc) {
  int lane = threadIdx.x & 31;
  int wid  = blockIdx.x * (blockDim.x >> 5) + (threadIdx.x >> 5);
  int nGroups = Nc >> 6;
  int total = ((M + 15) >> 4) * nGroups;
  if (wid >= total) return;
  int mt = wid / nGroups, ng = wid - mt * nGroups;
  int mbase = mt << 4, nbase = ng << 6;
  int lm = lane & 15, lhi = lane >> 4;
  int kba = lhi << 4;   // A-fragment segment base (flat 16 K per half-wave)
  int kbb = lhi << 4;   // B-fragment segment base (flat 16 K per half-wave)
  int row = mbase + lm; if (row >= M) row = M - 1;

  v8f acc[4] = {};
  for (int k0 = 0; k0 < K; k0 += 32) {
    const unsigned short* xr = inp + (size_t)row * K + k0 + kba;
    FragA a;
    a.q[0] = *(const uint4*)xr;
    a.q[1] = *(const uint4*)(xr + 8);
#pragma unroll
    for (int t = 0; t < 4; ++t) {
      int n = nbase + (t << 4) + lm;
      const unsigned short* wr = Wt + (size_t)n * K + k0 + kbb;
      FragA b;
      b.q[0] = *(const uint4*)wr;
      b.q[1] = *(const uint4*)(wr + 8);
      acc[t] = __builtin_amdgcn_wmma_f32_16x16x32_bf16(false, a.v, false, b.v,
                                                       (short)0, acc[t], false, false);
    }
  }
#pragma unroll
  for (int t = 0; t < 4; ++t) {
#pragma unroll
    for (int r = 0; r < 8; ++r) {
      int m = mbase + r + (lhi << 3);
      if (m < M)
        outp[(size_t)m * Nc + nbase + (t << 4) + lm] = f2bf(acc[t][r]);
    }
  }
}

// ---------- message passing: agg[dst] += bf2f(P[src]) * c_src[src] ----------
__global__ void k_scatter(const int* __restrict__ src, const int* __restrict__ dst,
                          const unsigned short* __restrict__ P, const float* __restrict__ csrc,
                          float* __restrict__ agg, int nE, int d) {
  int chunks = d >> 2;
  long long t = (long long)blockIdx.x * blockDim.x + threadIdx.x;
  if (t >= (long long)nE * chunks) return;
  int e = (int)(t / chunks);
  int f = ((int)(t - (long long)e * chunks)) << 2;
  int s = src[e], dd = dst[e];
  float cs = csrc[s];
  ushort4 pv = *(const ushort4*)(P + (size_t)s * d + f);
  float* ap = agg + (size_t)dd * d + f;
  atomicAdd(ap + 0, bf2f(pv.x) * cs);
  atomicAdd(ap + 1, bf2f(pv.y) * cs);
  atomicAdd(ap + 2, bf2f(pv.z) * cs);
  atomicAdd(ap + 3, bf2f(pv.w) * cs);
}

// h = (relu?)(agg * c_dst + bias) -> bf16
__global__ void k_finalize(const float* __restrict__ agg, const float* __restrict__ cdst,
                           const float* __restrict__ bias, unsigned short* __restrict__ h,
                           long long nTot, int d, int doRelu) {
  long long i = (long long)blockIdx.x * blockDim.x + threadIdx.x;
  if (i >= nTot) return;
  int node = (int)(i / d);
  int f = (int)(i - (long long)node * d);
  float v = agg[i] * cdst[node] + bias[f];
  if (doRelu) v = fmaxf(v, 0.0f);
  h[i] = f2bf(v);
}

// ---------- edge MLP decoder: 16 edges per wave, z = relu(cat[h_s,h_d]@Wp1+b), s = z@Wp2+b2 ----
__global__ void k_edge_mlp(const int* __restrict__ srcI, const int* __restrict__ dstI,
                           const unsigned short* __restrict__ H,     // [N x 64] bf16
                           const unsigned short* __restrict__ Wp1t,  // [64 x 128] bf16
                           const float* __restrict__ bp1, const float* __restrict__ Wp2,
                           const float* __restrict__ bp2, float* __restrict__ out, int nEdges) {
  int lane = threadIdx.x & 31;
  int wid  = blockIdx.x * (blockDim.x >> 5) + (threadIdx.x >> 5);
  int ntiles = (nEdges + 15) >> 4;
  if (wid >= ntiles) return;
  int ebase = wid << 4;
  int lm = lane & 15, lhi = lane >> 4;
  int er = ebase + lm; if (er >= nEdges) er = nEdges - 1;
  int s = srcI[er], dd = dstI[er];
  int kba = lhi << 4, kbb = lhi << 4;

  v8f acc[4] = {};
#pragma unroll
  for (int ks = 0; ks < 4; ++ks) {              // K = 128 concat dim
    const unsigned short* hp = H + (size_t)((ks < 2) ? s : dd) * 64;
    int fb = ((ks & 1) << 5) + kba;             // feature offset inside 64-wide half
    FragA a;
    a.q[0] = *(const uint4*)(hp + fb);
    a.q[1] = *(const uint4*)(hp + fb + 8);
    int k0 = ks << 5;
#pragma unroll
    for (int t = 0; t < 4; ++t) {               // N = 64 hidden
      int n = (t << 4) + lm;
      const unsigned short* wr = Wp1t + (size_t)n * 128 + k0 + kbb;
      FragA b;
      b.q[0] = *(const uint4*)wr;
      b.q[1] = *(const uint4*)(wr + 8);
      acc[t] = __builtin_amdgcn_wmma_f32_16x16x32_bf16(false, a.v, false, b.v,
                                                       (short)0, acc[t], false, false);
    }
  }
  float w2[4], bb[4];
#pragma unroll
  for (int t = 0; t < 4; ++t) { int n = (t << 4) + lm; bb[t] = bp1[n]; w2[t] = Wp2[n]; }
  float b2 = bp2[0];
#pragma unroll
  for (int r = 0; r < 8; ++r) {
    float sum = 0.0f;
#pragma unroll
    for (int t = 0; t < 4; ++t) {
      float z = fmaxf(acc[t][r] + bb[t], 0.0f);
      sum += z * w2[t];
    }
    sum += __shfl_xor(sum, 1, 32);
    sum += __shfl_xor(sum, 2, 32);
    sum += __shfl_xor(sum, 4, 32);
    sum += __shfl_xor(sum, 8, 32);
    if (lm == 0) {
      int m = ebase + r + (lhi << 3);
      if (m < nEdges) out[m] = sum + b2;
    }
  }
}

// ---------- launch ----------
extern "C" void kernel_launch(void* const* d_in, const int* in_sizes, int n_in,
                              void* d_out, int out_size, void* d_ws, size_t ws_size,
                              hipStream_t stream) {
  const float* x   = (const float*)d_in[0];
  const float* W0  = (const float*)d_in[1];
  const float* b0  = (const float*)d_in[2];
  const float* W1  = (const float*)d_in[3];
  const float* b1  = (const float*)d_in[4];
  const float* Wp1 = (const float*)d_in[5];
  const float* bp1 = (const float*)d_in[6];
  const float* Wp2 = (const float*)d_in[7];
  const float* bp2 = (const float*)d_in[8];
  const int* e0s = (const int*)d_in[9];
  const int* e0d = (const int*)d_in[10];
  const int* e1s = (const int*)d_in[11];
  const int* e1d = (const int*)d_in[12];
  const int* ps  = (const int*)d_in[13];
  const int* pd  = (const int*)d_in[14];
  const int* ns  = (const int*)d_in[15];
  const int* nd  = (const int*)d_in[16];
  float* out = (float*)d_out;
  (void)n_in; (void)out_size; (void)ws_size;

  const int DIN = 512, DH = 256, DO = 64;
  const int Nn  = in_sizes[0] / DIN;
  const int E0  = in_sizes[9];
  const int E1  = in_sizes[11];
  const int EP_ = in_sizes[13];

  char* ws = (char*)d_ws;
  size_t off = 0;
  auto alloc = [&](size_t bytes) -> char* {
    char* p = ws + off;
    off = (off + bytes + 255) & ~(size_t)255;
    return p;
  };
  float* c0s = (float*)alloc((size_t)Nn * 4);
  float* c0d = (float*)alloc((size_t)Nn * 4);
  float* c1s = (float*)alloc((size_t)Nn * 4);
  float* c1d = (float*)alloc((size_t)Nn * 4);
  unsigned short* W0t  = (unsigned short*)alloc((size_t)DIN * DH * 2);
  unsigned short* W1t  = (unsigned short*)alloc((size_t)DH * DO * 2);
  unsigned short* Wp1t = (unsigned short*)alloc((size_t)2 * DO * DO * 2);
  unsigned short* P    = (unsigned short*)alloc((size_t)Nn * DH * 2); // proj out (reused layer2)
  // Xbf (bf16 copy of x, Nn*DIN*2 bytes) aliases AGG (Nn*DH*4 bytes) — identical size,
  // Xbf is dead after GEMM1, AGG is born at the k_zero_f32 that follows it.
  char* aggRegion      = alloc((size_t)Nn * DH * 4);
  unsigned short* Xbf  = (unsigned short*)aggRegion;
  float*          AGG  = (float*)aggRegion;
  unsigned short* H0   = (unsigned short*)alloc((size_t)Nn * DH * 2);
  unsigned short* H1   = (unsigned short*)alloc((size_t)Nn * DO * 2);

  const int TB = 256;
  auto g1 = [&](long long n) { return dim3((unsigned)((n + TB - 1) / TB)); };

  // 1. degree -> normalization coefficients
  k_zero_f32<<<g1(Nn), TB, 0, stream>>>(c0s, Nn);
  k_zero_f32<<<g1(Nn), TB, 0, stream>>>(c0d, Nn);
  k_zero_f32<<<g1(Nn), TB, 0, stream>>>(c1s, Nn);
  k_zero_f32<<<g1(Nn), TB, 0, stream>>>(c1d, Nn);
  k_count_deg<<<g1(E0), TB, 0, stream>>>(e0s, c0s, E0);
  k_count_deg<<<g1(E0), TB, 0, stream>>>(e0d, c0d, E0);
  k_count_deg<<<g1(E1), TB, 0, stream>>>(e1s, c1s, E1);
  k_count_deg<<<g1(E1), TB, 0, stream>>>(e1d, c1d, E1);
  k_norm_coef<<<g1(Nn), TB, 0, stream>>>(c0s, Nn);
  k_norm_coef<<<g1(Nn), TB, 0, stream>>>(c0d, Nn);
  k_norm_coef<<<g1(Nn), TB, 0, stream>>>(c1s, Nn);
  k_norm_coef<<<g1(Nn), TB, 0, stream>>>(c1d, Nn);

  // 2. weights -> bf16 transposed [Nc x K]; x -> bf16 (streaming, into AGG alias)
  k_convT<<<g1((long long)DIN * DH), TB, 0, stream>>>(W0, W0t, DIN, DH);
  k_convT<<<g1((long long)DH * DO), TB, 0, stream>>>(W1, W1t, DH, DO);
  k_convT<<<g1((long long)2 * DO * DO), TB, 0, stream>>>(Wp1, Wp1t, 2 * DO, DO);
  k_conv_bf16_vec4<<<g1((long long)Nn * DIN / 4), TB, 0, stream>>>(x, Xbf,
                                                                   (long long)Nn * DIN / 4);

  // 3. layer 1: project (WMMA), scatter, normalize+bias+relu
  {
    int waves = ((Nn + 15) / 16) * (DH / 64);
    k_gemm_wmma<<<(waves + 7) / 8, 256, 0, stream>>>(Xbf, W0t, P, Nn, DIN, DH);
  }
  k_zero_f32<<<g1((long long)Nn * DH), TB, 0, stream>>>(AGG, (long long)Nn * DH);
  k_scatter<<<g1((long long)E0 * (DH / 4)), TB, 0, stream>>>(e0s, e0d, P, c0s, AGG, E0, DH);
  k_finalize<<<g1((long long)Nn * DH), TB, 0, stream>>>(AGG, c0d, b0, H0,
                                                        (long long)Nn * DH, DH, 1);

  // 4. layer 2
  {
    int waves = ((Nn + 15) / 16) * (DO / 64);
    k_gemm_wmma<<<(waves + 7) / 8, 256, 0, stream>>>(H0, W1t, P, Nn, DH, DO);
  }
  k_zero_f32<<<g1((long long)Nn * DO), TB, 0, stream>>>(AGG, (long long)Nn * DO);
  k_scatter<<<g1((long long)E1 * (DO / 4)), TB, 0, stream>>>(e1s, e1d, P, c1s, AGG, E1, DO);
  k_finalize<<<g1((long long)Nn * DO), TB, 0, stream>>>(AGG, c1d, b1, H1,
                                                        (long long)Nn * DO, DO, 0);

  // 5. edge MLP decode (WMMA), pos then neg
  {
    int ntiles = (EP_ + 15) / 16;
    int blocks = (ntiles + 3) / 4;
    k_edge_mlp<<<blocks, 128, 0, stream>>>(ps, pd, H1, Wp1t, bp1, Wp2, bp2, out, EP_);
    k_edge_mlp<<<blocks, 128, 0, stream>>>(ns, nd, H1, Wp1t, bp1, Wp2, bp2, out + EP_, EP_);
  }
}
